// TrajectoryPredictorLSTM_82051055223370
// MI455X (gfx1250) — compile-verified, matrix operand order checked by reading, not run on hardware
//
#include <hip/hip_runtime.h>

// TrajectoryPredictorLSTM on gfx1250 (MI455X).
// Strategy: batch-tiled recurrent GEMMs via V_WMMA_F32_16X16X32_F16.
//   - workgroup = 32 batch rows (two M=16 WMMA tiles), 8 waves (wave32)
//   - wave w owns gate columns [16w, 16w+16) of the 128 LSTM gates (N-tiles)
//   - recurrent weights (f32->f16) live as register-resident B fragments
//   - h0/h1 tiles live in LDS (f16); c0/c1 live in registers per fixed lane map
//   - layer1 pre-activation = WMMA(h0_new, Wih1) + WMMA(h1, Whh1) chained
//   - tiny K=2 input term + biases + sigmoid/tanh + FC head done in VALU
//   - activations use v_exp_f32 + v_rcp_f32 (no IEEE divide sequences)

typedef __attribute__((ext_vector_type(16))) _Float16 v16h;
typedef __attribute__((ext_vector_type(8)))  float    v8f;

#define BT   32   // batch rows per workgroup (two 16-row WMMA M-tiles)
#define HDIM 32   // hidden size
#define TENC 11
#define TDEC 80

union AFrag { v16h v; unsigned u[8]; _Float16 h[16]; };

__device__ __forceinline__ v8f wmma_f16(v16h a, v16h b, v8f c) {
  // 8 args: (neg_a, A, neg_b, B, c_mod, C, reuse_a, reuse_b)
  return __builtin_amdgcn_wmma_f32_16x16x32_f16(false, a, false, b, (short)0, c,
                                                false, false);
}

// A fragment: 16x32 f16 (M x K). ISA layout:
// lanes 0-15: M=lane, K pairs {0,2,4,6} then {16,18,20,22}
// lanes 16-31: M=lane-16, K pairs {8..14} then {24..30}
__device__ __forceinline__ v16h load_a_frag(const _Float16* sh, int mtile, int lane) {
  AFrag f;
  const unsigned* p = (const unsigned*)sh;   // 2 f16 per u32, row stride 16 u32
  const int m  = mtile * 16 + (lane & 15);
  const int kb = (lane >> 4) * 8;
#pragma unroll
  for (int j = 0; j < 8; ++j) {
    const int k0 = kb + ((j < 4) ? (2 * j) : (8 + 2 * j)); // {0,2,4,6,16,18,20,22}+kb
    f.u[j] = p[m * 16 + (k0 >> 1)];
  }
  return f.v;
}

// B fragment: 32x16 f16 (K x N), B[k][n] = W[n][k], W row-major (128 x 32) f32.
// Mirrors A layout with N across lanes.
__device__ __forceinline__ v16h load_b_frag(const float* __restrict__ W, int wid, int lane) {
  AFrag f;
  const int n  = wid * 16 + (lane & 15);
  const int kb = (lane >> 4) * 8;
#pragma unroll
  for (int j = 0; j < 8; ++j) {
    const int k0 = kb + ((j < 4) ? (2 * j) : (8 + 2 * j));
    f.h[2 * j]     = (_Float16)W[n * HDIM + k0];
    f.h[2 * j + 1] = (_Float16)W[n * HDIM + k0 + 1];
  }
  return f.v;
}

// C/D layout: VGPR r, lanes 0-15 -> (M=r, N=lane); lanes 16-31 -> (M=8+r, N=lane-16)
__device__ __forceinline__ void scatter_acc(float* sg, v8f acc, int wid, int mtile, int lane) {
  const int n  = wid * 16 + (lane & 15);
  const int mb = mtile * 16 + ((lane >> 4) << 3);
#pragma unroll
  for (int r = 0; r < 8; ++r) sg[(mb + r) * 128 + n] = acc[r];
}

// Fast activations: v_exp_f32 + v_rcp_f32; saturate correctly at +/-inf.
__device__ __forceinline__ float rcp_fast(float v) { return __builtin_amdgcn_rcpf(v); }
__device__ __forceinline__ float sigm(float v)  { return rcp_fast(1.0f + __expf(-v)); }
__device__ __forceinline__ float tanh_(float v) { return 1.0f - 2.0f * rcp_fast(__expf(2.0f * v) + 1.0f); }

__device__ __forceinline__ float cellupd(float gi, float gf, float gg, float go, float& c) {
  c = sigm(gf) * c + sigm(gi) * tanh_(gg);
  return sigm(go) * tanh_(c);
}

__global__ __launch_bounds__(256) void traj_lstm_wmma(
    const float* __restrict__ x,
    const float* __restrict__ eWih0, const float* __restrict__ eWhh0,
    const float* __restrict__ ebih0, const float* __restrict__ ebhh0,
    const float* __restrict__ eWih1, const float* __restrict__ eWhh1,
    const float* __restrict__ ebih1, const float* __restrict__ ebhh1,
    const float* __restrict__ dWih0, const float* __restrict__ dWhh0,
    const float* __restrict__ dbih0, const float* __restrict__ dbhh0,
    const float* __restrict__ dWih1, const float* __restrict__ dWhh1,
    const float* __restrict__ dbih1, const float* __restrict__ dbhh1,
    const float* __restrict__ fcW,   const float* __restrict__ fcb,
    float* __restrict__ out)
{
  __shared__ _Float16 sh0[BT * HDIM];     // layer0 hidden, f16 row-major [b][k]
  __shared__ _Float16 sh1[BT * HDIM];     // layer1 hidden
  __shared__ float    sgates[BT * 128];   // pre-activations for one layer
  __shared__ float    sx[BT * TENC * 2];  // input tile
  __shared__ float    spred[BT * 2];      // decoder feedback
  __shared__ float    sfcw[2 * HDIM];     // fc weights

  const int tid  = threadIdx.x;
  const int lane = tid & 31;
  const int wid  = tid >> 5;                       // N-tile owner (0..7)
  const long long b0 = (long long)blockIdx.x * BT; // batch base

  // --- one-time loads -------------------------------------------------------
  for (int i = tid; i < BT * TENC * 2; i += 256)
    sx[i] = x[b0 * (TENC * 2) + i];                // x tile is contiguous
  if (tid < 2 * HDIM) sfcw[tid] = fcW[tid];
  for (int i = tid; i < BT * HDIM; i += 256) {
    sh0[i] = (_Float16)0.0f;
    sh1[i] = (_Float16)0.0f;
  }

  // register-resident B fragments (weights are L2-resident; loaded once)
  const v16h fWhh0e = load_b_frag(eWhh0, wid, lane);
  const v16h fWih1e = load_b_frag(eWih1, wid, lane);
  const v16h fWhh1e = load_b_frag(eWhh1, wid, lane);
  const v16h fWhh0d = load_b_frag(dWhh0, wid, lane);
  const v16h fWih1d = load_b_frag(dWih1, wid, lane);
  const v16h fWhh1d = load_b_frag(dWhh1, wid, lane);

  // elementwise lane map: this lane updates (b,h) for b = eb+8p (p=0..3), h = eh
  const int eh = tid & (HDIM - 1);
  const int eb = tid >> 5;
  float be0[4], be1[4], bd0[4], bd1[4], wxe[8], wxd[8];
#pragma unroll
  for (int q = 0; q < 4; ++q) {
    be0[q] = ebih0[32 * q + eh] + ebhh0[32 * q + eh];
    be1[q] = ebih1[32 * q + eh] + ebhh1[32 * q + eh];
    bd0[q] = dbih0[32 * q + eh] + dbhh0[32 * q + eh];
    bd1[q] = dbih1[32 * q + eh] + dbhh1[32 * q + eh];
    wxe[2 * q]     = eWih0[(32 * q + eh) * 2];
    wxe[2 * q + 1] = eWih0[(32 * q + eh) * 2 + 1];
    wxd[2 * q]     = dWih0[(32 * q + eh) * 2];
    wxd[2 * q + 1] = dWih0[(32 * q + eh) * 2 + 1];
  }
  const float fcbj = (tid < 64) ? fcb[tid >> 5] : 0.0f;

  float c0r[4] = {0.f, 0.f, 0.f, 0.f};
  float c1r[4] = {0.f, 0.f, 0.f, 0.f};
  __syncthreads();

  // --- encoder: 11 steps ----------------------------------------------------
  for (int t = 0; t < TENC; ++t) {
    // layer0: h0 @ Whh0e
    v8f g0 = {0.f,0.f,0.f,0.f,0.f,0.f,0.f,0.f};
    v8f g1 = {0.f,0.f,0.f,0.f,0.f,0.f,0.f,0.f};
    g0 = wmma_f16(load_a_frag(sh0, 0, lane), fWhh0e, g0);
    g1 = wmma_f16(load_a_frag(sh0, 1, lane), fWhh0e, g1);
    scatter_acc(sgates, g0, wid, 0, lane);
    scatter_acc(sgates, g1, wid, 1, lane);
    __syncthreads();
#pragma unroll
    for (int p = 0; p < 4; ++p) {
      const int b = eb + 8 * p;
      const float x0 = sx[b * 22 + 2 * t], x1 = sx[b * 22 + 2 * t + 1];
      const float gi = sgates[b * 128       + eh] + be0[0] + x0 * wxe[0] + x1 * wxe[1];
      const float gf = sgates[b * 128 +  32 + eh] + be0[1] + x0 * wxe[2] + x1 * wxe[3];
      const float gg = sgates[b * 128 +  64 + eh] + be0[2] + x0 * wxe[4] + x1 * wxe[5];
      const float go = sgates[b * 128 +  96 + eh] + be0[3] + x0 * wxe[6] + x1 * wxe[7];
      sh0[b * HDIM + eh] = (_Float16)cellupd(gi, gf, gg, go, c0r[p]);
    }
    __syncthreads();
    // layer1: h0_new @ Wih1e + h1 @ Whh1e (two chained WMMAs per tile)
    v8f u0 = {0.f,0.f,0.f,0.f,0.f,0.f,0.f,0.f};
    v8f u1 = {0.f,0.f,0.f,0.f,0.f,0.f,0.f,0.f};
    u0 = wmma_f16(load_a_frag(sh0, 0, lane), fWih1e, u0);
    u1 = wmma_f16(load_a_frag(sh0, 1, lane), fWih1e, u1);
    u0 = wmma_f16(load_a_frag(sh1, 0, lane), fWhh1e, u0);
    u1 = wmma_f16(load_a_frag(sh1, 1, lane), fWhh1e, u1);
    scatter_acc(sgates, u0, wid, 0, lane);
    scatter_acc(sgates, u1, wid, 1, lane);
    __syncthreads();
#pragma unroll
    for (int p = 0; p < 4; ++p) {
      const int b = eb + 8 * p;
      const float gi = sgates[b * 128       + eh] + be1[0];
      const float gf = sgates[b * 128 +  32 + eh] + be1[1];
      const float gg = sgates[b * 128 +  64 + eh] + be1[2];
      const float go = sgates[b * 128 +  96 + eh] + be1[3];
      sh1[b * HDIM + eh] = (_Float16)cellupd(gi, gf, gg, go, c1r[p]);
    }
    __syncthreads();
  }

  // decoder seed: dec_in0 = x[:, -1, :]
  if (tid < BT * 2) spred[tid] = sx[(tid >> 1) * 22 + 20 + (tid & 1)];
  __syncthreads();

  // --- decoder: 80 steps ----------------------------------------------------
  for (int t = 0; t < TDEC; ++t) {
    v8f g0 = {0.f,0.f,0.f,0.f,0.f,0.f,0.f,0.f};
    v8f g1 = {0.f,0.f,0.f,0.f,0.f,0.f,0.f,0.f};
    g0 = wmma_f16(load_a_frag(sh0, 0, lane), fWhh0d, g0);
    g1 = wmma_f16(load_a_frag(sh0, 1, lane), fWhh0d, g1);
    scatter_acc(sgates, g0, wid, 0, lane);
    scatter_acc(sgates, g1, wid, 1, lane);
    __syncthreads();
#pragma unroll
    for (int p = 0; p < 4; ++p) {
      const int b = eb + 8 * p;
      const float x0 = spred[b * 2], x1 = spred[b * 2 + 1];
      const float gi = sgates[b * 128       + eh] + bd0[0] + x0 * wxd[0] + x1 * wxd[1];
      const float gf = sgates[b * 128 +  32 + eh] + bd0[1] + x0 * wxd[2] + x1 * wxd[3];
      const float gg = sgates[b * 128 +  64 + eh] + bd0[2] + x0 * wxd[4] + x1 * wxd[5];
      const float go = sgates[b * 128 +  96 + eh] + bd0[3] + x0 * wxd[6] + x1 * wxd[7];
      sh0[b * HDIM + eh] = (_Float16)cellupd(gi, gf, gg, go, c0r[p]);
    }
    __syncthreads();
    v8f u0 = {0.f,0.f,0.f,0.f,0.f,0.f,0.f,0.f};
    v8f u1 = {0.f,0.f,0.f,0.f,0.f,0.f,0.f,0.f};
    u0 = wmma_f16(load_a_frag(sh0, 0, lane), fWih1d, u0);
    u1 = wmma_f16(load_a_frag(sh0, 1, lane), fWih1d, u1);
    u0 = wmma_f16(load_a_frag(sh1, 0, lane), fWhh1d, u0);
    u1 = wmma_f16(load_a_frag(sh1, 1, lane), fWhh1d, u1);
    scatter_acc(sgates, u0, wid, 0, lane);
    scatter_acc(sgates, u1, wid, 1, lane);
    __syncthreads();
#pragma unroll
    for (int p = 0; p < 4; ++p) {
      const int b = eb + 8 * p;
      const float gi = sgates[b * 128       + eh] + bd1[0];
      const float gf = sgates[b * 128 +  32 + eh] + bd1[1];
      const float gg = sgates[b * 128 +  64 + eh] + bd1[2];
      const float go = sgates[b * 128 +  96 + eh] + bd1[3];
      sh1[b * HDIM + eh] = (_Float16)cellupd(gi, gf, gg, go, c1r[p]);
    }
    __syncthreads();
    // fc head: pred = h1 @ fcW.T + fcb ; write output + feedback
    if (tid < 64) {
      const int b = tid & 31, j = tid >> 5;
      float acc = fcbj;
#pragma unroll
      for (int k = 0; k < HDIM; ++k)
        acc += (float)sh1[b * HDIM + k] * sfcw[j * HDIM + k];
      spred[b * 2 + j] = acc;
      out[(b0 + b) * (TDEC * 2) + 2 * t + j] = acc;
    }
    __syncthreads();
  }
}

extern "C" void kernel_launch(void* const* d_in, const int* in_sizes, int n_in,
                              void* d_out, int out_size, void* d_ws, size_t ws_size,
                              hipStream_t stream) {
  const float* x     = (const float*)d_in[0];
  const float* eWih0 = (const float*)d_in[1];
  const float* eWhh0 = (const float*)d_in[2];
  const float* ebih0 = (const float*)d_in[3];
  const float* ebhh0 = (const float*)d_in[4];
  const float* eWih1 = (const float*)d_in[5];
  const float* eWhh1 = (const float*)d_in[6];
  const float* ebih1 = (const float*)d_in[7];
  const float* ebhh1 = (const float*)d_in[8];
  const float* dWih0 = (const float*)d_in[9];
  const float* dWhh0 = (const float*)d_in[10];
  const float* dbih0 = (const float*)d_in[11];
  const float* dbhh0 = (const float*)d_in[12];
  const float* dWih1 = (const float*)d_in[13];
  const float* dWhh1 = (const float*)d_in[14];
  const float* dbih1 = (const float*)d_in[15];
  const float* dbhh1 = (const float*)d_in[16];
  const float* fcW   = (const float*)d_in[17];
  const float* fcb   = (const float*)d_in[18];
  float* out = (float*)d_out;

  const int B = in_sizes[0] / (TENC * 2);  // 131072
  const int grid = B / BT;                 // 4096 workgroups, 8 waves each

  traj_lstm_wmma<<<grid, 256, 0, stream>>>(
      x, eWih0, eWhh0, ebih0, ebhh0, eWih1, eWhh1, ebih1, ebhh1,
      dWih0, dWhh0, dbih0, dbhh0, dWih1, dWhh1, dbih1, dbhh1,
      fcW, fcb, out);
}
